// ROIHead_52742198395510
// MI455X (gfx1250) — matrix operand order repaired and checked by reading, not compiled, not run on hardware
//
#include <hip/hip_runtime.h>
#include <hip/hip_bf16.h>
#include <math.h>

// ---------------- constants (fixed by the reference) ----------------
#define N_PROP 1000
#define NPAD   1024          // 32 * 32  (M padded so 32x32 wave tiles fit exactly)
#define CIN    256
#define K1     12544         // 256 * 7 * 7  (= 128 * 98)
#define FCD    1024          // (= 128 * 8)
#define NC     21
#define NREG   84
#define FH     64
#define NCAND  20000         // 1000 * 20 (classes 1..20)
#define NSORT  32768
#define MNMS   1024
#define NTOPK  100
#define BBOXCLIP 4.135166556742356f  // log(1000/16)

typedef __attribute__((ext_vector_type(16))) __bf16 v16bf;
typedef __attribute__((ext_vector_type(8)))  __bf16 v8bf;
typedef __attribute__((ext_vector_type(8)))  float  v8f;

__device__ __forceinline__ unsigned short f2bf_bits(float f) {
    unsigned u = __float_as_uint(f);
    u += 0x7FFFu + ((u >> 16) & 1u);          // round-to-nearest-even
    return (unsigned short)(u >> 16);
}

// ------------- 1. transpose + fp32->bf16 convert:  WT[n][k] = bf16(W[k][n]) -------------
__global__ void k_transpose_bf16(const float* __restrict__ W,
                                 unsigned short* __restrict__ WT,
                                 int Kdim, int Ncol) {
    long long i = (long long)blockIdx.x * blockDim.x + threadIdx.x;
    long long total = (long long)Kdim * Ncol;
    if (i >= total) return;
    int n = (int)(i / Kdim);
    int k = (int)(i % Kdim);
    WT[i] = f2bf_bits(W[(long long)k * Ncol + n]);
}

// ------------- 2. ROI max-pool -> bf16 activation rows -------------
__global__ void k_roipool(const float* __restrict__ feat,   // [256][64][64]
                          const float* __restrict__ props,  // [1000][4]
                          unsigned short* __restrict__ Xp)  // [1024][12544]
{
    int n = blockIdx.x;
    int c = threadIdx.x;                       // 256 threads = one channel each
    unsigned short* row = Xp + (size_t)n * K1;
    if (n >= N_PROP) {                         // zero pad rows 1000..1023
        for (int j = c; j < K1; j += 256) row[j] = 0;
        return;
    }
    const float scale = 0.0625f;               // 2^round(log2(64/1024))
    float bx1 = props[n*4+0], by1 = props[n*4+1];
    float bx2 = props[n*4+2], by2 = props[n*4+3];
    int x1 = (int)rintf(bx1*scale), y1 = (int)rintf(by1*scale);
    int x2 = (int)rintf(bx2*scale), y2 = (int)rintf(by2*scale);
    float bw = fmaxf((float)(x2-x1+1), 1.0f) / 7.0f;
    float bh = fmaxf((float)(y2-y1+1), 1.0f) / 7.0f;
    const float* fc = feat + (size_t)c * FH * FH;
    for (int ph = 0; ph < 7; ++ph) {
        int hs = min(max((int)floorf(ph*bh) + y1, 0), FH);
        int he = min(max((int)ceilf((ph+1)*bh) + y1, 0), FH);
        for (int pw = 0; pw < 7; ++pw) {
            int ws_ = min(max((int)floorf(pw*bw) + x1, 0), FH);
            int we_ = min(max((int)ceilf((pw+1)*bw) + x1, 0), FH);
            float m = -INFINITY;
            for (int h = hs; h < he; ++h)
                for (int w = ws_; w < we_; ++w)
                    m = fmaxf(m, fc[h*FH + w]);
            float v = (he > hs && we_ > ws_) ? m : 0.0f;
            row[c*49 + ph*7 + pw] = f2bf_bits(v);
        }
    }
}

// ------------- 3/4. bf16 WMMA GEMM, 2x2 register-blocked: O = relu(A @ B^T + bias) ----
// A:  [NPAD x KTOT] bf16 row-major,  BT: [FCD x KTOT] bf16 (B pre-transposed).
// One wave owns a 32x32 output block = 4 accumulators; each A frag is reused by
// 2 WMMAs and each B frag by 2 WMMAs -> 2 b128 loads per WMMA instead of 3.
// Outer K loop steps 128 elems (256B = 2 cachelines/stream); unconditional
// WGP-scope prefetches stage L2 lines into WGP$ one outer-iteration group ahead,
// with zero branches in the hot loop.
template <int KTOT>
__global__ void k_gemm_bf16(const unsigned short* __restrict__ A,
                            const unsigned short* __restrict__ BT,
                            const float* __restrict__ bias,
                            unsigned short* __restrict__ Obf,   // may be null
                            float* __restrict__ Of)             // may be null
{
    int lane = threadIdx.x & 31;
    int wave = threadIdx.x >> 5;
    int mb = blockIdx.x * 32;                  // M block base (0..992 step 32)
    int nb = (blockIdx.y * 8 + wave) * 32;     // N block base (0..992 step 32)
    int hi = lane >> 4;
    int lm = lane & 15;
    const unsigned short* a0 = A  + (size_t)(mb + lm)      * KTOT + hi*16;
    const unsigned short* a1 = A  + (size_t)(mb + 16 + lm) * KTOT + hi*16;
    const unsigned short* b0 = BT + (size_t)(nb + lm)      * KTOT + hi*16;
    const unsigned short* b1 = BT + (size_t)(nb + 16 + lm) * KTOT + hi*16;
    v8f acc00 = {0.f,0.f,0.f,0.f,0.f,0.f,0.f,0.f};
    v8f acc01 = acc00, acc10 = acc00, acc11 = acc00;
    for (int k0 = 0; k0 < KTOT; k0 += 128) {
        // stage next group's lines from L2 into WGP$ (speculative; OOB dropped)
        __builtin_prefetch(a0 + k0 + 128, 0, 3);
        __builtin_prefetch(a0 + k0 + 192, 0, 3);
        __builtin_prefetch(a1 + k0 + 128, 0, 3);
        __builtin_prefetch(a1 + k0 + 192, 0, 3);
        __builtin_prefetch(b0 + k0 + 128, 0, 3);
        __builtin_prefetch(b0 + k0 + 192, 0, 3);
        __builtin_prefetch(b1 + k0 + 128, 0, 3);
        __builtin_prefetch(b1 + k0 + 192, 0, 3);
#pragma unroll
        for (int kk = k0; kk < k0 + 128; kk += 32) {
            union { v16bf v; v8bf h[2]; } fa0, fa1;
            fa0.h[0] = *reinterpret_cast<const v8bf*>(a0 + kk);
            fa0.h[1] = *reinterpret_cast<const v8bf*>(a0 + kk + 8);
            fa1.h[0] = *reinterpret_cast<const v8bf*>(a1 + kk);
            fa1.h[1] = *reinterpret_cast<const v8bf*>(a1 + kk + 8);
            v16bf fb0 = *reinterpret_cast<const v16bf*>(b0 + kk);
            v16bf fb1 = *reinterpret_cast<const v16bf*>(b1 + kk);
            acc00 = __builtin_amdgcn_wmma_f32_16x16x32_bf16(false, fa0.v, false, fb0, (short)0, acc00, false, false);
            acc01 = __builtin_amdgcn_wmma_f32_16x16x32_bf16(false, fa0.v, false, fb1, (short)0, acc01, false, false);
            acc10 = __builtin_amdgcn_wmma_f32_16x16x32_bf16(false, fa1.v, false, fb0, (short)0, acc10, false, false);
            acc11 = __builtin_amdgcn_wmma_f32_16x16x32_bf16(false, fa1.v, false, fb1, (short)0, acc11, false, false);
        }
    }
    // epilogue: C/D layout per tile: N = lane%16, M = vgpr + 8*(lane/16)
    float bc0 = bias[nb + lm];
    float bc1 = bias[nb + 16 + lm];
#pragma unroll
    for (int r = 0; r < 8; ++r) {
        int r0 = mb + r + 8*hi;
        int r1 = mb + 16 + r + 8*hi;
        float v00 = fmaxf(acc00[r] + bc0, 0.0f);
        float v01 = fmaxf(acc01[r] + bc1, 0.0f);
        float v10 = fmaxf(acc10[r] + bc0, 0.0f);
        float v11 = fmaxf(acc11[r] + bc1, 0.0f);
        if (Obf) {
            Obf[(size_t)r0 * FCD + nb + lm]      = f2bf_bits(v00);
            Obf[(size_t)r0 * FCD + nb + 16 + lm] = f2bf_bits(v01);
            Obf[(size_t)r1 * FCD + nb + lm]      = f2bf_bits(v10);
            Obf[(size_t)r1 * FCD + nb + 16 + lm] = f2bf_bits(v11);
        }
        if (Of) {
            Of[(size_t)r0 * FCD + nb + lm]      = v00;
            Of[(size_t)r0 * FCD + nb + 16 + lm] = v01;
            Of[(size_t)r1 * FCD + nb + lm]      = v10;
            Of[(size_t)r1 * FCD + nb + 16 + lm] = v11;
        }
    }
}

// ------------- 5. classification + regression heads (skinny, plain fp32) -------------
__global__ void k_heads(const float* __restrict__ X3,
                        const float* __restrict__ Wc, const float* __restrict__ bc,
                        const float* __restrict__ Wr, const float* __restrict__ br,
                        float* __restrict__ cls, float* __restrict__ reg)
{
    int n = blockIdx.x;
    int t = threadIdx.x;                   // 128 threads, 105 used
    if (t >= NC + NREG) return;
    const float* x = X3 + (size_t)n * FCD;
    float acc = 0.f;
    if (t < NC) {
        for (int k = 0; k < FCD; ++k) acc += x[k] * Wc[(size_t)k*NC + t];
        cls[n*NC + t] = acc + bc[t];
    } else {
        int r = t - NC;
        for (int k = 0; k < FCD; ++k) acc += x[k] * Wr[(size_t)k*NREG + r];
        reg[n*NREG + r] = acc + br[r];
    }
}

// ------------- 6a. key init -------------
__global__ void k_initkeys(unsigned long long* keys) {
    int i = blockIdx.x * blockDim.x + threadIdx.x;
    if (i < NSORT) keys[i] = 0ull;
}

// ------------- 6b. softmax + box decode + validity keys -------------
__global__ void k_decode(const float* __restrict__ props,
                         const float* __restrict__ cls,
                         const float* __restrict__ reg,
                         const int* __restrict__ pimh, const int* __restrict__ pimw,
                         float* __restrict__ boxes, float* __restrict__ scores,
                         unsigned long long* __restrict__ keys)
{
    __shared__ float sh[NC];
    __shared__ float red[2];
    int n = blockIdx.x;
    int t = threadIdx.x;                   // 32 threads
    if (t < NC) sh[t] = cls[n*NC + t];
    __syncthreads();
    if (t == 0) {
        float m = sh[0];
        for (int i = 1; i < NC; ++i) m = fmaxf(m, sh[i]);
        red[0] = m;
    }
    __syncthreads();
    float m = red[0];
    if (t < NC) sh[t] = __expf(sh[t] - m);
    __syncthreads();
    if (t == 0) {
        float s = 0.f;
        for (int i = 0; i < NC; ++i) s += sh[i];
        red[1] = s;
    }
    __syncthreads();
    if (t >= 1 && t < NC) {
        float imw = (float)pimw[0], imh = (float)pimh[0];
        float prob = sh[t] / red[1];
        float ax1 = props[n*4+0], ay1 = props[n*4+1];
        float ax2 = props[n*4+2], ay2 = props[n*4+3];
        float w = ax2 - ax1, h = ay2 - ay1;
        float cx = ax1 + 0.5f*w, cy = ay1 + 0.5f*h;
        const float* d = reg + (size_t)n*NREG + t*4;
        float dx = d[0], dy = d[1];
        float dw = fminf(d[2], BBOXCLIP), dh = fminf(d[3], BBOXCLIP);
        float pcx = dx*w + cx, pcy = dy*h + cy;
        float pw = __expf(dw)*w, ph = __expf(dh)*h;
        float x1 = fminf(fmaxf(pcx - 0.5f*pw, 0.f), imw);
        float y1 = fminf(fmaxf(pcy - 0.5f*ph, 0.f), imh);
        float x2 = fminf(fmaxf(pcx + 0.5f*pw, 0.f), imw);
        float y2 = fminf(fmaxf(pcy + 0.5f*ph, 0.f), imh);
        int idx = n*20 + (t-1);
        boxes[idx*4+0] = x1; boxes[idx*4+1] = y1;
        boxes[idx*4+2] = x2; boxes[idx*4+3] = y2;
        scores[idx] = prob;
        bool valid = (prob > 0.01f) && ((x2-x1) >= 16.0f) && ((y2-y1) >= 16.0f);
        keys[idx] = valid
            ? ((((unsigned long long)__float_as_uint(prob)) << 32) | (unsigned)idx)
            : 0ull;
    }
}

// ------------- 7. bitonic sort step (descending on u64 = score||idx) -------------
__global__ void k_bitonic(unsigned long long* keys, int j, int k) {
    int i = blockIdx.x * blockDim.x + threadIdx.x;
    int ixj = i ^ j;
    if (ixj > i) {
        unsigned long long a = keys[i], b = keys[ixj];
        bool descRegion = ((i & k) == 0);
        bool doSwap = descRegion ? (a < b) : (a > b);
        if (doSwap) { keys[i] = b; keys[ixj] = a; }
    }
}

// ------------- 8. single-block NMS on top-1024 + top-100 compaction -------------
__global__ void k_nms(const unsigned long long* __restrict__ keys,
                      const float* __restrict__ boxes,
                      const int* __restrict__ pimh, const int* __restrict__ pimw,
                      float* __restrict__ out)
{
    __shared__ float bx[MNMS][4];
    __shared__ float sc[MNMS];
    __shared__ int   lb[MNMS];
    __shared__ int   ix[MNMS];
    __shared__ unsigned char keep[MNMS];
    int t = threadIdx.x;
    float imw = (float)pimw[0], imh = (float)pimh[0];
    unsigned long long key = keys[t];
    int idx = (int)(key & 0xffffffffull);
    float score = __uint_as_float((unsigned)(key >> 32));
    bool valid = (key != 0ull);
    int label = (idx % 20) + 1;
    float off = (float)label * (imw + imh);    // class-separating offset
    bx[t][0] = boxes[idx*4+0] + off;
    bx[t][1] = boxes[idx*4+1] + off;
    bx[t][2] = boxes[idx*4+2] + off;
    bx[t][3] = boxes[idx*4+3] + off;
    sc[t] = valid ? score : -1.0f;
    lb[t] = label;
    ix[t] = idx;
    keep[t] = valid ? 1 : 0;
    __syncthreads();
    for (int i = 0; i < MNMS; ++i) {
        if (keep[i] && t > i && keep[t]) {
            float iw = fmaxf(fminf(bx[i][2], bx[t][2]) - fmaxf(bx[i][0], bx[t][0]), 0.f);
            float ih = fmaxf(fminf(bx[i][3], bx[t][3]) - fmaxf(bx[i][1], bx[t][1]), 0.f);
            float inter = iw * ih;
            float a1 = fmaxf(bx[i][2]-bx[i][0], 0.f) * fmaxf(bx[i][3]-bx[i][1], 0.f);
            float a2 = fmaxf(bx[t][2]-bx[t][0], 0.f) * fmaxf(bx[t][3]-bx[t][1], 0.f);
            float uni = fmaxf(a1 + a2 - inter, 1e-6f);
            if (inter / uni > 0.5f) keep[t] = 0;
        }
        __syncthreads();
    }
    // init outputs: boxes[0..399]=0, scores[400..499]=0, labels[500..599]=-1
    if (t < NTOPK) {
        out[t*4+0] = 0.f; out[t*4+1] = 0.f; out[t*4+2] = 0.f; out[t*4+3] = 0.f;
        out[400 + t] = 0.f;
        ((int*)out)[500 + t] = -1;
    }
    __syncthreads();
    if (t == 0) {                               // keys already sorted desc -> compact
        int r = 0;
        for (int i = 0; i < MNMS && r < NTOPK; ++i) {
            if (keep[i]) {
                int id = ix[i];
                out[r*4+0] = boxes[id*4+0];
                out[r*4+1] = boxes[id*4+1];
                out[r*4+2] = boxes[id*4+2];
                out[r*4+3] = boxes[id*4+3];
                out[400 + r] = sc[i];
                ((int*)out)[500 + r] = lb[i];
                ++r;
            }
        }
    }
}

// ---------------- host-side launch ----------------
extern "C" void kernel_launch(void* const* d_in, const int* in_sizes, int n_in,
                              void* d_out, int out_size, void* d_ws, size_t ws_size,
                              hipStream_t stream) {
    const float* feat  = (const float*)d_in[0];
    const float* props = (const float*)d_in[1];
    const float* W6 = (const float*)d_in[2];
    const float* b6 = (const float*)d_in[3];
    const float* W7 = (const float*)d_in[4];
    const float* b7 = (const float*)d_in[5];
    const float* Wc = (const float*)d_in[6];
    const float* bc = (const float*)d_in[7];
    const float* Wr = (const float*)d_in[8];
    const float* br = (const float*)d_in[9];
    const int* imh = (const int*)d_in[10];
    const int* imw = (const int*)d_in[11];

    char* ws = (char*)d_ws;
    size_t o = 0;
    auto carve = [&](size_t bytes) -> void* {
        o = (o + 255) & ~(size_t)255;
        void* p = ws + o;
        o += bytes;
        return p;
    };
    unsigned short* W6T = (unsigned short*)carve((size_t)FCD * K1 * 2);   // 25.7 MB
    unsigned short* W7T = (unsigned short*)carve((size_t)FCD * FCD * 2);  //  2.1 MB
    unsigned short* Xp  = (unsigned short*)carve((size_t)NPAD * K1 * 2);  // 25.7 MB
    unsigned short* X2  = (unsigned short*)carve((size_t)NPAD * FCD * 2); //  2.1 MB
    float* X3      = (float*)carve((size_t)NPAD * FCD * 4);               //  4.2 MB
    float* clsbuf  = (float*)carve((size_t)N_PROP * NC * 4);
    float* regbuf  = (float*)carve((size_t)N_PROP * NREG * 4);
    float* boxesc  = (float*)carve((size_t)NCAND * 4 * 4);
    float* scoresc = (float*)carve((size_t)NCAND * 4);
    unsigned long long* keys = (unsigned long long*)carve((size_t)NSORT * 8);

    // 1. weight convert/transpose to bf16 K-contiguous
    {
        long long tot = (long long)K1 * FCD;
        k_transpose_bf16<<<(unsigned)((tot + 255) / 256), 256, 0, stream>>>(W6, W6T, K1, FCD);
        long long tot7 = (long long)FCD * FCD;
        k_transpose_bf16<<<(unsigned)((tot7 + 255) / 256), 256, 0, stream>>>(W7, W7T, FCD, FCD);
    }
    // 2. ROI pool -> bf16 rows (padded to 1024)
    k_roipool<<<NPAD, 256, 0, stream>>>(feat, props, Xp);
    // 3. FC6 (WMMA bf16, K=12544, 2x2 blocked) -> bf16
    k_gemm_bf16<K1><<<dim3(32, 4), 256, 0, stream>>>(Xp, W6T, b6, X2, nullptr);
    // 4. FC7 (WMMA bf16, K=1024, 2x2 blocked) -> fp32
    k_gemm_bf16<FCD><<<dim3(32, 4), 256, 0, stream>>>(X2, W7T, b7, nullptr, X3);
    // 5. heads
    k_heads<<<N_PROP, 128, 0, stream>>>(X3, Wc, bc, Wr, br, clsbuf, regbuf);
    // 6. softmax/decode/keys
    k_initkeys<<<NSORT / 256, 256, 0, stream>>>(keys);
    k_decode<<<N_PROP, 32, 0, stream>>>(props, clsbuf, regbuf, imh, imw,
                                        boxesc, scoresc, keys);
    // 7. bitonic sort (descending) == top_k(1024)
    for (int k = 2; k <= NSORT; k <<= 1)
        for (int j = k >> 1; j > 0; j >>= 1)
            k_bitonic<<<NSORT / 256, 256, 0, stream>>>(keys, j, k);
    // 8. NMS + top-100 output
    k_nms<<<1, MNMS, 0, stream>>>(keys, boxesc, imh, imw, (float*)d_out);
}
